// CrossConvLayerV2_72189810312030
// MI455X (gfx1250) — matrix operand order repaired and checked by reading, not compiled, not run on hardware
//
#include <hip/hip_runtime.h>

typedef __attribute__((ext_vector_type(16))) _Float16 v16h;
typedef __attribute__((ext_vector_type(8)))  float    v8f;

#define NPTS      4096
#define NCENT     256
#define M_PROBES  26
#define NFEAT     16
#define PT_DIM    19      // 3 coords + 16 feats
#define KCOEFF    0.1f
#define TWO_PI    6.2831853071795864f
#define NWAVES    8       // waves per workgroup
#define CHUNKS_PW (NPTS / (NWAVES * 32))   // chunks of 32 points per wave = 16

// probe offset m (0..25), radius 3.0, per the reference _make_probes()
__device__ __forceinline__ void probe_offset(int m, float& px, float& py, float& pz) {
    float a, b;
    if (m < 24) {
        int j = m >> 3, i = m & 7;
        a = ((float)j * 0.125f - 0.125f) * TWO_PI;
        b = ((float)i * 0.125f + ((float)j - 1.0f) * 0.0625f) * TWO_PI;
    } else {
        a = (m == 24 ? -0.25f : 0.25f) * TWO_PI;
        b = 0.0f;
    }
    float sa = sinf(a), ca = cosf(a), sb = sinf(b), cb = cosf(b);
    px = sa * 3.0f;
    py = ca * cb * 3.0f;
    pz = ca * sb * 3.0f;
}

// One workgroup per (b, l): 8 waves, each streams 512 points in chunks of 32,
// accumulating agg[32 x 16] with v_wmma_f32_16x16x32_f16 (two 16-row m-tiles).
// Inner eval uses |x-p|^2 + c = (|x|^2 + |p|^2 + c) - 2 p.x  -> 1 add + 3 FMA + rcp.
__global__ __launch_bounds__(NWAVES * 32)
void crossconv_fused_kernel(const float* __restrict__ points,
                            const float* __restrict__ centers,
                            const float* __restrict__ W,       // 416 x 256
                            const float* __restrict__ bias,    // 256
                            float* __restrict__ out)           // (B*l) x 256
{
    __shared__ float    s_coord[NWAVES][4][32];   // x, y, z, |x|^2 per staged point
    __shared__ _Float16 s_featT[NWAVES][16][32];  // feats transposed [feat][point]
    __shared__ float    s_red[NWAVES][2][8][32];  // cross-wave accumulator reduction
    __shared__ float    s_flat[M_PROBES * NFEAT]; // 416 flattened agg row

    const int tid  = threadIdx.x;
    const int lane = tid & 31;
    const int w    = tid >> 5;          // wave id 0..NWAVES-1
    const int bl   = blockIdx.x;        // b*256 + l
    const int b    = bl >> 8;

    const float cx = centers[bl * 3 + 0];
    const float cy = centers[bl * 3 + 1];
    const float cz = centers[bl * 3 + 2];

    // Per-lane probe constants for its two m-rows (tile0: m, tile1: m+16).
    const int mrow = lane & 15;
    float p0x, p0y, p0z;
    probe_offset(mrow, p0x, p0y, p0z);
    p0x += cx; p0y += cy; p0z += cz;
    const float n2p0x = -2.0f * p0x, n2p0y = -2.0f * p0y, n2p0z = -2.0f * p0z;
    const float base0 = p0x * p0x + p0y * p0y + p0z * p0z + KCOEFF;

    float n2p1x = 0.f, n2p1y = 0.f, n2p1z = 0.f;
    float base1 = 1e30f;   // invalid rows: d ~ 1e30 -> rcp ~ 1e-30 -> f16 rounds to 0
    if (mrow + 16 < M_PROBES) {
        float p1x, p1y, p1z;
        probe_offset(mrow + 16, p1x, p1y, p1z);
        p1x += cx; p1y += cy; p1z += cz;
        n2p1x = -2.0f * p1x; n2p1y = -2.0f * p1y; n2p1z = -2.0f * p1z;
        base1 = p1x * p1x + p1y * p1y + p1z * p1z + KCOEFF;
    }

    // ISA VGPR-layout bases (wave32):
    // A (16x32 f16): lane holds row M=lane&15; K base = (lane>>4)*8.
    // B (32x16 f16): lane holds col N=lane&15; K base = (lane>>4)*8 (same k order as A).
    const int kbaseA = (lane >> 4) << 3;
    const int kbaseB = (lane >> 4) << 4;
    const int nfeat  = lane & 15;

    v8f acc0 = {};   // m rows 0..15  x f 0..15, f32 accum (of 1/d, scaled later)
    v8f acc1 = {};   // m rows 16..31 (only 16..25 valid)

    const float* ptbase = points + (size_t)b * NPTS * PT_DIM;

    for (int ch = 0; ch < CHUNKS_PW; ++ch) {
        // ---- stage this wave's chunk of 32 points into LDS ----
        const int p = ((w * CHUNKS_PW + ch) << 5) + lane;
        const float* pp = ptbase + (size_t)p * PT_DIM;
        const float x = pp[0], y = pp[1], z = pp[2];
        s_coord[w][0][lane] = x;
        s_coord[w][1][lane] = y;
        s_coord[w][2][lane] = z;
        s_coord[w][3][lane] = x * x + y * y + z * z;
        #pragma unroll
        for (int ff = 0; ff < NFEAT; ++ff)
            s_featT[w][ff][lane] = (_Float16)pp[3 + ff];
        __syncthreads();

        // ---- build A (1/denominator values) and B (feats) fragments ----
        v16h amat0, amat1, bmat;
        #pragma unroll
        for (int r = 0; r < 8; ++r) {
            const int kA = ((r >> 2) << 4) + kbaseA + ((r & 3) << 1);
            #pragma unroll
            for (int h = 0; h < 2; ++h) {
                const int k  = kA + h;
                const float px_ = s_coord[w][0][k];
                const float py_ = s_coord[w][1][k];
                const float pz_ = s_coord[w][2][k];
                const float tt  = s_coord[w][3][k];

                float d0 = tt + base0;
                d0 = fmaf(n2p0x, px_, d0);
                d0 = fmaf(n2p0y, py_, d0);
                d0 = fmaf(n2p0z, pz_, d0);
                amat0[2 * r + h] = (_Float16)__builtin_amdgcn_rcpf(d0);

                float d1 = tt + base1;
                d1 = fmaf(n2p1x, px_, d1);
                d1 = fmaf(n2p1y, py_, d1);
                d1 = fmaf(n2p1z, pz_, d1);
                amat1[2 * r + h] = (_Float16)__builtin_amdgcn_rcpf(d1);
            }
            const int kB = ((r >> 2) << 4) + kbaseA + ((r & 3) << 1);
            bmat[2 * r]     = s_featT[w][nfeat][kB];
            bmat[2 * r + 1] = s_featT[w][nfeat][kB + 1];
        }

        // D = A x B + C (f16 in, f32 accum); co-executes with the VALU above
        acc0 = __builtin_amdgcn_wmma_f32_16x16x32_f16(false, amat0, false, bmat,
                                                      (short)0, acc0, false, false);
        acc1 = __builtin_amdgcn_wmma_f32_16x16x32_f16(false, amat1, false, bmat,
                                                      (short)0, acc1, false, false);
        __syncthreads();
    }

    // ---- reduce the per-wave accumulators through LDS ----
    #pragma unroll
    for (int r = 0; r < 8; ++r) {
        s_red[w][0][r][lane] = acc0[r];
        s_red[w][1][r][lane] = acc1[r];
    }
    __syncthreads();

    const float scale = KCOEFF / (float)NPTS;   // fold COEFF and 1/n here
    for (int idx = tid; idx < 512; idx += NWAVES * 32) {
        const int t  = idx >> 8;
        const int r  = (idx >> 5) & 7;
        const int ln = idx & 31;
        float s = 0.f;
        #pragma unroll
        for (int ww = 0; ww < NWAVES; ++ww)
            s += s_red[ww][t][r][ln];
        // 32-bit C/D layout: m = 16*t + r + 8*(ln>>4), f = ln&15
        const int m = t * 16 + r + ((ln >> 4) << 3);
        const int f = ln & 15;
        if (m < M_PROBES)
            s_flat[m * NFEAT + f] = s * scale;
    }
    __syncthreads();

    // ---- final 1x416 @ 416x256 GEMV (W stays L2-resident across all 512 blocks) ----
    for (int c = tid; c < 256; c += NWAVES * 32) {
        float accv = bias[c];
        #pragma unroll 4
        for (int k = 0; k < M_PROBES * NFEAT; ++k)
            accv = fmaf(s_flat[k], W[k * 256 + c], accv);
        out[(size_t)bl * 256 + c] = accv;
    }
}

extern "C" void kernel_launch(void* const* d_in, const int* in_sizes, int n_in,
                              void* d_out, int out_size, void* d_ws, size_t ws_size,
                              hipStream_t stream) {
    const float* points  = (const float*)d_in[0];  // (2, 4096, 19)
    const float* centers = (const float*)d_in[1];  // (2, 256, 3)
    const float* W       = (const float*)d_in[2];  // (416, 256)
    const float* bias    = (const float*)d_in[3];  // (256,)
    float* out = (float*)d_out;                    // (2, 256, 256)

    crossconv_fused_kernel<<<dim3(2 * NCENT), dim3(NWAVES * 32), 0, stream>>>(
        points, centers, W, bias, out);
}
